// TopKRouter_87067577024915
// MI455X (gfx1250) — matrix operand order repaired and checked by reading, not compiled, not run on hardware
//
#include <hip/hip_runtime.h>
#include <hip/hip_bf16.h>

// TopK MoE router for MI455X (gfx1250, wave32).
// logits = x(32768x2048,f32) @ gate_w(64x2048,f32)^T via bf16 WMMA (f32 accum),
// then per-token top-2 + softmax + one-hot mask, fused in one kernel.
//
// Layout: 256 WGs x 128 threads (4 wave32s). Each wave computes 32 tokens
// (2 M-tiles) x 64 experts (4 N-tiles) -> 8 independent v_wmma accumulators,
// so each B fragment load/convert feeds two WMMAs and the scheduler has
// independent matrix ops to cover the bf16 WMMA->VALU hazard slots.

#define NTOK       32768
#define DIM        2048
#define NEXP       64
#define TOK_PER_WG 128
#define THREADS    128
#define LDS_STRIDE 65   // 64 + 1 pad -> conflict-free per-token scans

typedef __attribute__((ext_vector_type(16))) __bf16 v16bf;
typedef __attribute__((ext_vector_type(8)))  float  v8f;

union BF16x16 { v16bf v; __bf16 e[16]; };
union F32x8   { v8f   v; float  e[8];  };

static __device__ __forceinline__ void f4pair_to_bf8(__bf16* dst, float4 u, float4 w) {
    dst[0] = (__bf16)u.x; dst[1] = (__bf16)u.y; dst[2] = (__bf16)u.z; dst[3] = (__bf16)u.w;
    dst[4] = (__bf16)w.x; dst[5] = (__bf16)w.y; dst[6] = (__bf16)w.z; dst[7] = (__bf16)w.w;
}

static __device__ __forceinline__ void load_a_frag(BF16x16& A, const float* xr,
                                                   int kb, int kA0, int kA1) {
    float4 a0 = *(const float4*)(xr + kb + kA0);
    float4 a1 = *(const float4*)(xr + kb + kA0 + 4);
    float4 a2 = *(const float4*)(xr + kb + kA1);
    float4 a3 = *(const float4*)(xr + kb + kA1 + 4);
    f4pair_to_bf8(A.e,     a0, a1);
    f4pair_to_bf8(A.e + 8, a2, a3);
}

__global__ __launch_bounds__(THREADS)
void topk_router_kernel(const float* __restrict__ x,
                        const float* __restrict__ gw,
                        float* __restrict__ out)
{
    __shared__ float s_logits[TOK_PER_WG * LDS_STRIDE];
    __shared__ int   s_i0[TOK_PER_WG];
    __shared__ int   s_i1[TOK_PER_WG];

    const int tid    = threadIdx.x;
    const int wave   = tid >> 5;          // 4 waves
    const int lane   = tid & 31;
    const int laneLo = lane & 15;
    const int laneHi = lane >> 4;         // 0 or 1

    // ---------------- Phase 1: GEMM logits via bf16 WMMA ----------------
    // Wave owns tokens [tokBase, tokBase+32): two 16-row M-tiles.
    const int tokBase = blockIdx.x * TOK_PER_WG + wave * 32;
    const float* xr0 = x + (size_t)(tokBase +      laneLo) * DIM;
    const float* xr1 = x + (size_t)(tokBase + 16 + laneLo) * DIM;

    // A fragment k-offsets within a 32-wide chunk (ISA 16-bit A 16x32 layout):
    // lane<16: K 0..7 then 16..23 ; lane>=16: K 8..15 then 24..31
    const int kA0 = laneHi * 8;
    const int kA1 = 16 + laneHi * 8;
    // B fragment: same K order as A (lane<16: K 0..7,16..23; lane>=16: K 8..15,24..31), col N=laneLo
    const int kB  = laneHi * 8;

    v8f acc[2][4];
    #pragma unroll
    for (int m = 0; m < 2; ++m)
        #pragma unroll
        for (int t = 0; t < 4; ++t) acc[m][t] = (v8f){};

    for (int kb = 0; kb < DIM; kb += 32) {
        BF16x16 A0, A1;
        load_a_frag(A0, xr0, kb, kA0, kA1);
        load_a_frag(A1, xr1, kb, kA0, kA1);
        if (kb + 64 < DIM) {
            __builtin_prefetch(xr0 + kb + 64, 0, 1);
            __builtin_prefetch(xr1 + kb + 64, 0, 1);
        }

        #pragma unroll
        for (int t = 0; t < 4; ++t) {
            // B: dims kB..kB+7 and kB+16..kB+23 of expert row (t*16 + laneLo), L2-resident.
            const float* wr = gw + (size_t)(t * 16 + laneLo) * DIM + kb + kB;
            float4 b0 = *(const float4*)(wr);
            float4 b1 = *(const float4*)(wr + 4);
            float4 b2 = *(const float4*)(wr + 16);
            float4 b3 = *(const float4*)(wr + 20);
            BF16x16 B;
            f4pair_to_bf8(B.e,     b0, b1);
            f4pair_to_bf8(B.e + 8, b2, b3);

            // Two independent WMMAs share this B fragment.
            acc[0][t] = __builtin_amdgcn_wmma_f32_16x16x32_bf16(
                            false, A0.v, false, B.v, (short)0, acc[0][t], false, false);
            acc[1][t] = __builtin_amdgcn_wmma_f32_16x16x32_bf16(
                            false, A1.v, false, B.v, (short)0, acc[1][t], false, false);
        }
    }

    // C layout: VGPR r -> M = laneHi*8 + r, N = laneLo. Scatter to padded LDS.
    #pragma unroll
    for (int m = 0; m < 2; ++m) {
        #pragma unroll
        for (int t = 0; t < 4; ++t) {
            F32x8 C; C.v = acc[m][t];
            #pragma unroll
            for (int r = 0; r < 8; ++r) {
                int row = wave * 32 + m * 16 + laneHi * 8 + r;
                s_logits[row * LDS_STRIDE + t * 16 + laneLo] = C.e[r];
            }
        }
    }
    __syncthreads();

    // ---------------- Phase 2: per-token top-2 + softmax ----------------
    {
        const float* row = &s_logits[tid * LDS_STRIDE];
        float b0 = -3.402823466e+38f, b1 = -3.402823466e+38f;
        int   i0 = 0, i1 = 0;
        #pragma unroll 8
        for (int e = 0; e < NEXP; ++e) {
            float v = row[e];
            if (v > b0)      { b1 = b0; i1 = i0; b0 = v; i0 = e; }
            else if (v > b1) { b1 = v;  i1 = e; }
        }
        float d   = __expf(b1 - b0);     // <= 1 since b1 <= b0
        float inv = 1.0f / (1.0f + d);
        const int tok = blockIdx.x * TOK_PER_WG + tid;
        float* w_out = out + (size_t)NTOK * NEXP;
        int*   i_out = (int*)(out + (size_t)NTOK * NEXP + (size_t)NTOK * 2);
        w_out[2 * tok + 0] = inv;        // softmax of [top1, top2]
        w_out[2 * tok + 1] = d * inv;
        i_out[2 * tok + 0] = i0;
        i_out[2 * tok + 1] = i1;
        s_i0[tid] = i0;
        s_i1[tid] = i1;
    }
    __syncthreads();

    // ---------------- Phase 3: coalesced one-hot mask write ----------------
    float* mask_out = out + (size_t)blockIdx.x * TOK_PER_WG * NEXP;
    for (int i = tid; i < TOK_PER_WG * NEXP; i += THREADS) {
        int tl = i >> 6;
        int e  = i & 63;
        mask_out[i] = (e == s_i0[tl] || e == s_i1[tl]) ? 1.0f : 0.0f;
    }
}

extern "C" void kernel_launch(void* const* d_in, const int* in_sizes, int n_in,
                              void* d_out, int out_size, void* d_ws, size_t ws_size,
                              hipStream_t stream) {
    (void)in_sizes; (void)n_in; (void)out_size; (void)d_ws; (void)ws_size;
    const float* x  = (const float*)d_in[0];
    const float* gw = (const float*)d_in[1];
    float* out = (float*)d_out;

    dim3 grid(NTOK / TOK_PER_WG);   // 256 workgroups
    dim3 block(THREADS);            // 4 wave32s per WG
    topk_router_kernel<<<grid, block, 0, stream>>>(x, gw, out);
}
